// BM25Encoder_27590869909670
// MI455X (gfx1250) — matrix operand order repaired and checked by reading, not compiled, not run on hardware
//
#include <hip/hip_runtime.h>
#include <hip/hip_bf16.h>
#include <math.h>

// ---------------------------------------------------------------------------
// BM25 encoder for MI455X (gfx1250, wave32, WMMA).
// Strategy: densify v into bf16 in a pre-swizzled WMMA-A fragment layout,
// swizzle W into the WMMA-B fragment layout, run a bf16 WMMA GEMM with f32
// accumulation (94.5 GFLOP). Both operands (123MB + 46MB) fit in the 192MB L2,
// so fragment loads go straight global->VGPR with no LDS staging.
// Workspace required: ~178 MB.
// ---------------------------------------------------------------------------

typedef __attribute__((ext_vector_type(16))) __bf16 v16bf;
typedef __attribute__((ext_vector_type(8)))  float  v8f;

#define VOCABN 30000
#define KP     30016          // vocab padded to multiple of 32
#define KSTEPS (KP / 32)      // 938 K-steps of 32
#define SEQL   512
#define DOUT   768
#define NT16   (DOUT / 16)    // 48 column tiles of 16

__device__ __forceinline__ unsigned short f2bf(float f) {
  union { float f; unsigned u; } v; v.f = f;
  unsigned r = v.u + 0x7FFFu + ((v.u >> 16) & 1u);   // round-to-nearest-even
  return (unsigned short)(r >> 16);
}

// Swizzle maps (CDNA5 ISA 7.12.2, 16-bit operands, wave32):
//  A (16x32, M x K): lane = (m%16) + 16*((kk>>3)&1); slot = (kk&7) + 8*(kk>>4)
//  B (32x16, K x N): lane = (n%16) + 16*((kk>>3)&1); slot = (kk&7) + 8*(kk>>4)
// Fragment for (tile, lane) = 16 contiguous halves (32 bytes).

// ---- kernel 1: per-doc TF / BM25 score / vocab-L2 norm --------------------
__global__ void k_scores(const int* __restrict__ ids_g, const int* __restrict__ mask_g,
                         int* __restrict__ uniq_tok, float* __restrict__ uniq_val,
                         float* __restrict__ bg, int* __restrict__ nnz_out) {
  __shared__ int   s_ids[SEQL];
  __shared__ float s_dlen, s_sumsq;
  __shared__ int   s_cnt;
  const int t = threadIdx.x, b = blockIdx.x;
  const int id = ids_g[b * SEQL + t];
  const int m  = mask_g[b * SEQL + t];
  const bool valid = (m == 1) && (id > 100) && (id < VOCABN);
  const int key = valid ? id : -(t + 1);      // unique sentinel for invalid
  s_ids[t] = key;
  if (t == 0) { s_dlen = 0.f; s_sumsq = 0.f; s_cnt = 0; }
  __syncthreads();
  if (valid) atomicAdd(&s_dlen, 1.0f);
  int cnt = 0, first = t;
  #pragma unroll 8
  for (int i = 0; i < SEQL; ++i) {            // broadcast read: conflict-free
    int v = s_ids[i];
    if (v == key) { cnt++; if (i < first) first = i; }
  }
  __syncthreads();
  const float dlen = s_dlen;
  const bool isfirst = valid && (first == t);
  float entry = 0.f; int pos = -1;
  if (isfirst) {
    float ln = fmaxf(1.0f + 0.75f * (dlen * 0.01f - 1.0f), 0.5f);
    float tf = (float)cnt;
    float score = tf * 2.2f / (tf + 1.2f * ln);     // (k1+1)=2.2, k1=1.2
    entry = score + 1e-10f;
    atomicAdd(&s_sumsq, entry * entry);
    pos = atomicAdd(&s_cnt, 1);
    uniq_tok[b * SEQL + pos] = id;
  }
  __syncthreads();
  const int nnz = s_cnt;
  const float sumsq = s_sumsq + (float)(VOCABN - nnz) * 1e-20f;  // dense 1e-10 floor
  const float inv = 1.0f / sqrtf(sumsq);
  if (isfirst) uniq_val[b * SEQL + pos] = entry * inv;
  if (t == 0) { bg[b] = 1e-10f * inv; nnz_out[b] = nnz; }
}

// ---- kernel 2: fill Vsw with per-doc background (bf16, swizzled layout) ---
// Vsw[m16][ks][lane][slot]; one 16B chunk (8 halves) per thread, fully coalesced.
__global__ void k_fill(unsigned short* __restrict__ Vsw, const float* __restrict__ bg,
                       long long total) {
  long long gid = (long long)blockIdx.x * blockDim.x + threadIdx.x;
  if (gid >= total) return;
  int l = (int)(gid & 63);                 // chunk within the 1KB (m16,ks) block
  int lane = l >> 1;
  long long blk = gid >> 6;                // m16*KSTEPS + ks
  int m16 = (int)(blk / KSTEPS);
  int doc = m16 * 16 + (lane & 15);
  unsigned short h = f2bf(bg[doc]);
  unsigned p = ((unsigned)h) | ((unsigned)h << 16);
  ((uint4*)Vsw)[gid] = make_uint4(p, p, p, p);
}

// ---- kernel 3: scatter normalized scores into swizzled slots --------------
__global__ void k_scatter(unsigned short* __restrict__ Vsw, const int* __restrict__ uniq_tok,
                          const float* __restrict__ uniq_val, const int* __restrict__ nnz) {
  int b = blockIdx.x, t = threadIdx.x;
  if (t >= nnz[b]) return;
  int tok = uniq_tok[b * SEQL + t];
  float v  = uniq_val[b * SEQL + t];
  int ks = tok >> 5, kk = tok & 31;
  int lane = (b & 15) + 16 * ((kk >> 3) & 1);
  int slot = (kk & 7) + 8 * (kk >> 4);
  long long idx = (((((long long)(b >> 4)) * KSTEPS + ks) * 32 + lane) << 4) + slot;
  Vsw[idx] = f2bf(v);
}

// ---- kernel 4: W (768 x 30000 f32) -> Wsw bf16 in WMMA-B layout -----------
__global__ void k_wswz(const float* __restrict__ W, unsigned short* __restrict__ Wsw,
                       long long total) {
  long long gid = (long long)blockIdx.x * blockDim.x + threadIdx.x;
  if (gid >= total) return;
  int t = (int)(gid % KP);                 // t fastest -> coalesced W reads
  int j = (int)(gid / KP);
  float w = (t < VOCABN) ? W[(long long)j * VOCABN + t] : 0.0f;  // zero K-pad
  int ks = t >> 5, kk = t & 31;
  int lane = (j & 15) + 16 * ((kk >> 3) & 1);
  int slot = (kk & 7) + 8 * (kk >> 4);
  long long idx = ((((long long)ks * NT16 + (j >> 4)) * 32 + lane) << 4) + slot;
  Wsw[idx] = f2bf(w);
}

// ---- kernel 5: bf16 WMMA GEMM, out[b][j] = sum_t V[b][t]*W[j][t] ----------
// Block = 4 waves (2M x 2N); wave tile = 32x64 (2x4 WMMA frags); WG tile 64x128.
__global__ __launch_bounds__(128) void k_gemm(const unsigned short* __restrict__ Vsw,
                                              const unsigned short* __restrict__ Wsw,
                                              float* __restrict__ out) {
  const int lane = threadIdx.x & 31;
  const int w = threadIdx.x >> 5;
  const int wm = w & 1, wn = w >> 1;
  const int m16b = blockIdx.x * 4 + wm * 2;     // 2 m16 tiles per wave
  const int n16b = blockIdx.y * 8 + wn * 4;     // 4 n16 tiles per wave

  v8f c[2][4];
  #pragma unroll
  for (int mi = 0; mi < 2; ++mi)
    #pragma unroll
    for (int ni = 0; ni < 4; ++ni) c[mi][ni] = (v8f)0.0f;

  for (int ks = 0; ks < KSTEPS; ++ks) {
    v16bf a[2], bf[4];
    #pragma unroll
    for (int mi = 0; mi < 2; ++mi) {
      long long idx = ((((long long)(m16b + mi) * KSTEPS + ks) * 32 + lane) << 4);
      a[mi] = *reinterpret_cast<const v16bf*>(Vsw + idx);      // 32B/lane
    }
    #pragma unroll
    for (int ni = 0; ni < 4; ++ni) {
      long long idx = ((((long long)ks * NT16 + (n16b + ni)) * 32 + lane) << 4);
      bf[ni] = *reinterpret_cast<const v16bf*>(Wsw + idx);     // 32B/lane
    }
    if (ks + 1 < KSTEPS) {  // global_prefetch_b8 for the next K-step
      __builtin_prefetch(Vsw + ((((long long)m16b * KSTEPS + ks + 1) * 32 + lane) << 4), 0, 1);
      __builtin_prefetch(Wsw + ((((long long)(ks + 1) * NT16 + n16b) * 32 + lane) << 4), 0, 1);
    }
    #pragma unroll
    for (int mi = 0; mi < 2; ++mi)
      #pragma unroll
      for (int ni = 0; ni < 4; ++ni)
        c[mi][ni] = __builtin_amdgcn_wmma_f32_16x16x32_bf16(
            false, a[mi], false, bf[ni], (short)0, c[mi][ni], false, false);
  }

  // C/D layout: lanes 0-15: N=lane, M=r; lanes 16-31: N=lane-16, M=8+r
  const int row8 = (lane >> 4) * 8;
  const int col  = lane & 15;
  #pragma unroll
  for (int mi = 0; mi < 2; ++mi)
    #pragma unroll
    for (int ni = 0; ni < 4; ++ni)
      #pragma unroll
      for (int r = 0; r < 8; ++r) {
        int doc = (m16b + mi) * 16 + row8 + r;
        int j   = (n16b + ni) * 16 + col;
        out[(long long)doc * DOUT + j] = c[mi][ni][r];
      }
}

// ---- kernel 6: row-wise L2 normalize of out (2048 x 768) ------------------
__global__ void k_norm(float* __restrict__ out) {
  __shared__ float s_sum;
  int b = blockIdx.x, t = threadIdx.x;      // 256 threads x 3 elems
  if (t == 0) s_sum = 0.f;
  __syncthreads();
  float v0 = out[(long long)b * DOUT + t];
  float v1 = out[(long long)b * DOUT + 256 + t];
  float v2 = out[(long long)b * DOUT + 512 + t];
  atomicAdd(&s_sum, v0 * v0 + v1 * v1 + v2 * v2);
  __syncthreads();
  float inv = 1.0f / sqrtf(s_sum);
  out[(long long)b * DOUT + t]       = v0 * inv;
  out[(long long)b * DOUT + 256 + t] = v1 * inv;
  out[(long long)b * DOUT + 512 + t] = v2 * inv;
}

// ---------------------------------------------------------------------------
extern "C" void kernel_launch(void* const* d_in, const int* in_sizes, int n_in,
                              void* d_out, int out_size, void* d_ws, size_t ws_size,
                              hipStream_t stream) {
  const int*   ids  = (const int*)d_in[0];
  const int*   mask = (const int*)d_in[1];
  const float* W    = (const float*)d_in[2];
  float* out = (float*)d_out;
  const int B = in_sizes[0] / SEQL;       // 2048
  const int M16 = B / 16;

  char* ws = (char*)d_ws;
  size_t off = 0;
  auto take = [&](size_t bytes) { size_t o = off; off = (off + bytes + 255) & ~(size_t)255; return o; };
  unsigned short* Vsw = (unsigned short*)(ws + take((size_t)B * KP * 2));   // ~123 MB
  unsigned short* Wsw = (unsigned short*)(ws + take((size_t)KP * DOUT * 2));// ~46 MB
  int*   tok = (int*)  (ws + take((size_t)B * SEQL * 4));
  float* val = (float*)(ws + take((size_t)B * SEQL * 4));
  float* bg  = (float*)(ws + take((size_t)B * 4));
  int*   nnz = (int*)  (ws + take((size_t)B * 4));

  k_scores<<<B, SEQL, 0, stream>>>(ids, mask, tok, val, bg, nnz);

  long long fillTotal = (long long)M16 * KSTEPS * 64;
  k_fill<<<(unsigned)((fillTotal + 255) / 256), 256, 0, stream>>>(Vsw, bg, fillTotal);

  k_scatter<<<B, SEQL, 0, stream>>>(Vsw, tok, val, nnz);

  long long wTotal = (long long)KP * DOUT;
  k_wswz<<<(unsigned)((wTotal + 255) / 256), 256, 0, stream>>>(W, Wsw, wTotal);

  dim3 grid(B / 64, DOUT / 128);          // 32 x 6 workgroups
  k_gemm<<<grid, 128, 0, stream>>>(Vsw, Wsw, out);

  k_norm<<<B, 256, 0, stream>>>(out);
}
